// RmMultiHeadAttention_549755814364
// MI455X (gfx1250) — matrix operand order repaired and checked by reading, not compiled, not run on hardware
//
#include <hip/hip_runtime.h>

// Problem constants (from reference)
#define B_    2
#define S_    2048
#define HID_  1024
#define H_    16
#define C_    64
#define MROWS (B_ * S_) // 4096

typedef __bf16 bf16_t;
typedef bf16_t v16bf __attribute__((ext_vector_type(16)));
typedef float  v8f   __attribute__((ext_vector_type(8)));
typedef unsigned int v4u __attribute__((ext_vector_type(4)));
typedef int v8i __attribute__((ext_vector_type(8)));
typedef int v4i __attribute__((ext_vector_type(4)));

#if __has_builtin(__builtin_amdgcn_tensor_load_to_lds)
#define HAVE_TDM 1
#else
#define HAVE_TDM 0
#endif

union FragU {
  v16bf v;
  uint4 q[2];
};

static __device__ __forceinline__ unsigned short f2bf(float f) {
  unsigned int u = __float_as_uint(f);
  u += 0x7fffu + ((u >> 16) & 1u); // round-to-nearest-even
  return (unsigned short)(u >> 16);
}

// Build a 16-element bf16 fragment from two 16-byte runs.
static __device__ __forceinline__ v16bf load_frag(const unsigned short* p0,
                                                  const unsigned short* p1) {
  FragU f;
  f.q[0] = *reinterpret_cast<const uint4*>(p0);
  f.q[1] = *reinterpret_cast<const uint4*>(p1);
  return f.v;
}

static __device__ __forceinline__ v8f wmma_bf16(v16bf a, v16bf b, v8f c) {
  return __builtin_amdgcn_wmma_f32_16x16x32_bf16(false, a, false, b, (short)0, c,
                                                 false, false);
}

#if HAVE_TDM
// Issue a TDM 2D tile load: tile_h rows x tile_w bf16 elems, global row stride
// row_stride elems, into LDS at lds_off. One instruction per wave (EXEC ignored).
// D# layout per CDNA5 ISA ch.8 (group0 128b, group1 256b; groups 2/3 zero = 2D).
static __device__ __forceinline__ void tdm_load_tile_bf16(
    unsigned lds_off, const unsigned short* gptr,
    unsigned tile_w, unsigned tile_h, unsigned row_stride) {
  unsigned long long ga = (unsigned long long)(size_t)gptr;
  v4u g0;
  g0[0] = 1u;                                   // count=1, user descriptor
  g0[1] = lds_off;                              // lds_addr (bytes)
  g0[2] = (unsigned)(ga & 0xffffffffu);         // global_addr[31:0]
  g0[3] = (unsigned)((ga >> 32) & 0x1ffffffu)   // global_addr[56:32]
          | (2u << 30);                         // type=2 ("image")
  v8i g1;
  g1[0] = (int)(1u << 16);                      // data_size=1 -> 2 bytes/elem
  g1[1] = (int)(tile_w << 16);                  // tensor_dim0[15:0]
  g1[2] = (int)((tile_w >> 16) | (tile_h << 16)); // dim0[31:16] | dim1[15:0]
  g1[3] = (int)((tile_h >> 16) | (tile_w << 16)); // dim1[31:16] | tile_dim0
  g1[4] = (int)tile_h;                          // tile_dim1 (tile_dim2=0)
  g1[5] = (int)row_stride;                      // tensor_dim0_stride[31:0]
  g1[6] = 0;                                    // stride0 hi | stride1 lo
  g1[7] = 0;
  v4i gz = {0, 0, 0, 0};
#if defined(__clang_major__) && (__clang_major__ >= 23)
  v8i gz8 = {0, 0, 0, 0, 0, 0, 0, 0};
  __builtin_amdgcn_tensor_load_to_lds(g0, g1, gz, gz, gz8, 0);
#else
  __builtin_amdgcn_tensor_load_to_lds(g0, g1, gz, gz, 0);
#endif
}
#endif

// ---------------------------------------------------------------------------
// Kernel 1: projections. E[mat] = X[mat] @ W[mat], output bf16 into workspace.
// Grid: (N/64, M/64, 3), block 128 (4 waves). 64x64 tile/WG, 32x32 per wave.
// ---------------------------------------------------------------------------
__global__ void __launch_bounds__(128)
proj_bf16_wmma(const float* __restrict__ q, const float* __restrict__ k,
               const float* __restrict__ v, const float* __restrict__ wq,
               const float* __restrict__ wk, const float* __restrict__ wv,
               unsigned short* __restrict__ ws) {
  const int mat = blockIdx.z;
  const float* X = (mat == 0) ? q : (mat == 1) ? k : v;
  const float* W = (mat == 0) ? wq : (mat == 1) ? wk : wv;
  unsigned short* E = ws + (size_t)mat * (size_t)MROWS * HID_;
  const int m0 = blockIdx.y * 64;
  const int n0 = blockIdx.x * 64;

  __shared__ __align__(16) unsigned short LX[64 * 32]; // [m][k] bf16
  __shared__ __align__(16) unsigned short LW[64 * 32]; // transposed: [n][k] bf16

  const int tid  = threadIdx.x;
  const int lane = tid & 31;
  const int wid  = tid >> 5;
  const int half = lane >> 4;
  const int ln   = lane & 15;
  const int moff = (wid & 1) * 32;
  const int noff = (wid >> 1) * 32;

  const v8f vzero = {0.f, 0.f, 0.f, 0.f, 0.f, 0.f, 0.f, 0.f};
  v8f acc[2][2];
  acc[0][0] = vzero; acc[0][1] = vzero; acc[1][0] = vzero; acc[1][1] = vzero;

  for (int k0 = 0; k0 < HID_; k0 += 32) {
    // Stage X tile 64x32 (f32 -> bf16), coalesced float4 reads.
    {
      const int r  = tid >> 1;
      const int cb = (tid & 1) * 16;
      const float* src = X + (size_t)(m0 + r) * HID_ + k0 + cb;
#pragma unroll
      for (int j = 0; j < 16; j += 4) {
        float4 f = *reinterpret_cast<const float4*>(src + j);
        unsigned short* d = &LX[r * 32 + cb + j];
        d[0] = f2bf(f.x); d[1] = f2bf(f.y); d[2] = f2bf(f.z); d[3] = f2bf(f.w);
      }
    }
    // Stage W tile 32x64 transposed into [n][k] so B-fragments are contiguous.
    {
      const int kk = tid >> 2;
      const int nb = (tid & 3) * 16;
      const float* src = W + (size_t)(k0 + kk) * HID_ + n0 + nb;
#pragma unroll
      for (int j = 0; j < 16; j += 4) {
        float4 f = *reinterpret_cast<const float4*>(src + j);
        LW[(nb + j + 0) * 32 + kk] = f2bf(f.x);
        LW[(nb + j + 1) * 32 + kk] = f2bf(f.y);
        LW[(nb + j + 2) * 32 + kk] = f2bf(f.z);
        LW[(nb + j + 3) * 32 + kk] = f2bf(f.w);
      }
    }
    __syncthreads();

    v16bf afr[2], bfr[2];
#pragma unroll
    for (int mi = 0; mi < 2; ++mi) {
      // A 16x32 bf16: lane runs K = 8*half..+7 and 16+8*half..+7
      const unsigned short* p = &LX[(moff + mi * 16 + ln) * 32 + 8 * half];
      afr[mi] = load_frag(p, p + 16);
    }
#pragma unroll
    for (int ni = 0; ni < 2; ++ni) {
      // B 32x16 bf16: lane = column, K = 8*half..+7 and 16+8*half..+7
      const unsigned short* p = &LW[(noff + ni * 16 + ln) * 32 + 8 * half];
      bfr[ni] = load_frag(p, p + 16);
    }
#pragma unroll
    for (int mi = 0; mi < 2; ++mi)
#pragma unroll
      for (int ni = 0; ni < 2; ++ni)
        acc[mi][ni] = wmma_bf16(afr[mi], bfr[ni], acc[mi][ni]);
    __syncthreads();
  }

  // Write bf16 result (C/D layout: row = r + 8*half, col = ln).
#pragma unroll
  for (int mi = 0; mi < 2; ++mi)
#pragma unroll
    for (int ni = 0; ni < 2; ++ni)
#pragma unroll
      for (int r = 0; r < 8; ++r) {
        int row = m0 + moff + mi * 16 + r + 8 * half;
        int col = n0 + noff + ni * 16 + ln;
        E[(size_t)row * HID_ + col] = f2bf(acc[mi][ni][r]);
      }
}

// ---------------------------------------------------------------------------
// Kernel 2: causal flash attention per (b, h, 64-query block).
// Block 128 (4 waves); wave w owns queries [q0+16w, q0+16w+16).
// K chunk staged via TDM (tensor_load_to_lds), V chunk staged transposed.
// ---------------------------------------------------------------------------
__global__ void __launch_bounds__(128)
attn_flash_wmma(const unsigned short* __restrict__ ws, float* __restrict__ out) {
  const int b  = blockIdx.z;
  const int h  = blockIdx.y;
  const int q0 = blockIdx.x * 64;

  const size_t matsz = (size_t)MROWS * HID_;
  const size_t base  = ((size_t)b * S_) * HID_ + (size_t)h * C_;
  const unsigned short* EQ = ws + base;              // emb_q, bf16, row stride HID_
  const unsigned short* EK = ws + matsz + base;      // emb_k
  const unsigned short* EV = ws + 2 * matsz + base;  // emb_v

  __shared__ __align__(16) unsigned short LK[32 * 64]; // [t][c] bf16
  __shared__ __align__(16) unsigned short LV[64 * 32]; // transposed [c][t] bf16
  __shared__ __align__(16) unsigned short LP[4 * 16 * 32]; // per-wave P scratch

  const int tid  = threadIdx.x;
  const int lane = tid & 31;
  const int w    = tid >> 5;
  const int half = lane >> 4;
  const int ln   = lane & 15;
  const int qb   = q0 + w * 16;

  // Q fragments for the whole kernel (contraction C=64 -> two 32-chunks).
  v16bf aq[2];
#pragma unroll
  for (int kc = 0; kc < 2; ++kc) {
    const unsigned short* p = EQ + (size_t)(qb + ln) * HID_ + kc * 32 + 8 * half;
    aq[kc] = load_frag(p, p + 16);
  }

  const v8f vzero = {0.f, 0.f, 0.f, 0.f, 0.f, 0.f, 0.f, 0.f};
  v8f o[4];
#pragma unroll
  for (int i = 0; i < 4; ++i) o[i] = vzero;
  float mrow[8], lrow[8];
#pragma unroll
  for (int r = 0; r < 8; ++r) { mrow[r] = -1e30f; lrow[r] = 0.f; }

  const int tc_end = q0 + 64;
  for (int tc = 0; tc < tc_end; tc += 32) {
    // ---- Stage K chunk [32][64] bf16 into LDS ----
#if HAVE_TDM
    if (w == 0) {
      tdm_load_tile_bf16((unsigned)(size_t)(void*)LK,
                         EK + (size_t)tc * HID_,
                         /*tile_w=*/64, /*tile_h=*/32, /*row_stride=*/HID_);
    }
#else
    {
      const int tt = tid >> 2;
      const int cb = (tid & 3) * 16;
      const unsigned short* src = EK + (size_t)(tc + tt) * HID_ + cb;
      uint4 u0 = *reinterpret_cast<const uint4*>(src);
      uint4 u1 = *reinterpret_cast<const uint4*>(src + 8);
      *reinterpret_cast<uint4*>(&LK[tt * 64 + cb])     = u0;
      *reinterpret_cast<uint4*>(&LK[tt * 64 + cb + 8]) = u1;
    }
#endif
    // ---- Stage V chunk transposed [c][t]; prefetch next chunk ----
    {
      const int tt = tid >> 2;
      const int cb = (tid & 3) * 16;
      const unsigned short* src = EV + (size_t)(tc + tt) * HID_ + cb;
      uint4 u0 = *reinterpret_cast<const uint4*>(src);
      uint4 u1 = *reinterpret_cast<const uint4*>(src + 8);
      unsigned uu[8] = {u0.x, u0.y, u0.z, u0.w, u1.x, u1.y, u1.z, u1.w};
#pragma unroll
      for (int j = 0; j < 8; ++j) {
        LV[(cb + 2 * j + 0) * 32 + tt] = (unsigned short)(uu[j] & 0xffffu);
        LV[(cb + 2 * j + 1) * 32 + tt] = (unsigned short)(uu[j] >> 16);
      }
      if (tc + 32 < tc_end) { // prefetch next iteration's K/V rows
        __builtin_prefetch(EK + (size_t)(tc + 32 + tt) * HID_ + cb, 0, 1);
        __builtin_prefetch(EV + (size_t)(tc + 32 + tt) * HID_ + cb, 0, 1);
      }
    }
#if HAVE_TDM
    if (w == 0) __builtin_amdgcn_s_wait_tensorcnt(0);
#endif
    __syncthreads();

    if (tc <= qb + 15) { // chunk intersects this wave's causal region
      // --- scores: 16x32 tile = two 16x16 WMMA tiles, contraction C=64 ---
      v8f sc[2];
      sc[0] = vzero; sc[1] = vzero;
#pragma unroll
      for (int tt = 0; tt < 2; ++tt)
#pragma unroll
        for (int kc = 0; kc < 2; ++kc) {
          const unsigned short* p = &LK[(tt * 16 + ln) * 64 + kc * 32 + 8 * half];
          v16bf bk = load_frag(p, p + 16);
          sc[tt] = wmma_bf16(aq[kc], bk, sc[tt]);
        }

      // --- scale + causal mask (post-softmax tril+renorm == causal softmax) ---
      float s[2][8];
#pragma unroll
      for (int tt = 0; tt < 2; ++tt)
#pragma unroll
        for (int r = 0; r < 8; ++r) {
          float val = sc[tt][r] * 0.125f; // 1/sqrt(64)
          int tg = tc + tt * 16 + ln;
          int rg = qb + r + 8 * half;
          s[tt][r] = (tg > rg) ? -1e30f : val;
        }

      // --- online softmax: row max over 16 lanes of each half-wave ---
      float bm[8];
#pragma unroll
      for (int r = 0; r < 8; ++r) {
        bm[r] = fmaxf(s[0][r], s[1][r]);
#pragma unroll
        for (int d = 1; d < 16; d <<= 1)
          bm[r] = fmaxf(bm[r], __shfl_xor(bm[r], d, 16));
      }
      float rs[8];
#pragma unroll
      for (int r = 0; r < 8; ++r) {
        float mn   = fmaxf(mrow[r], bm[r]);
        float corr = __expf(mrow[r] - mn);
        mrow[r] = mn;
        float p0 = __expf(s[0][r] - mn);
        float p1 = __expf(s[1][r] - mn);
        s[0][r] = p0; s[1][r] = p1;
        rs[r]   = p0 + p1;
        lrow[r] *= corr;
#pragma unroll
        for (int cg = 0; cg < 4; ++cg) o[cg][r] *= corr;
      }
#pragma unroll
      for (int r = 0; r < 8; ++r) {
#pragma unroll
        for (int d = 1; d < 16; d <<= 1) rs[r] += __shfl_xor(rs[r], d, 16);
        lrow[r] += rs[r];
      }

      // --- reshape P (C-layout) -> A-fragment via per-wave LDS scratch ---
#pragma unroll
      for (int tt = 0; tt < 2; ++tt)
#pragma unroll
        for (int r = 0; r < 8; ++r)
          LP[w * 512 + (r + 8 * half) * 32 + tt * 16 + ln] = f2bf(s[tt][r]);
      __threadfence_block(); // order ds stores before ds loads (same wave)
      const unsigned short* pp = &LP[w * 512 + ln * 32 + 8 * half];
      v16bf pa = load_frag(pp, pp + 16);

      // --- accumulate P @ V (contraction = 32 keys, one WMMA per 16 cols) ---
#pragma unroll
      for (int cg = 0; cg < 4; ++cg) {
        const unsigned short* pv = &LV[(cg * 16 + ln) * 32 + 8 * half];
        v16bf bv = load_frag(pv, pv + 16);
        o[cg] = wmma_bf16(pa, bv, o[cg]);
      }
    }
    __syncthreads();
  }

  // Epilogue: out[b][row][h*64 + c] = acc / l
  float* OB = out + base;
#pragma unroll
  for (int r = 0; r < 8; ++r) {
    float inv = 1.0f / lrow[r];
    int row = qb + r + 8 * half;
#pragma unroll
    for (int cg = 0; cg < 4; ++cg)
      OB[(size_t)row * HID_ + cg * 16 + ln] = o[cg][r] * inv;
  }
}

// ---------------------------------------------------------------------------
extern "C" void kernel_launch(void* const* d_in, const int* in_sizes, int n_in,
                              void* d_out, int out_size, void* d_ws, size_t ws_size,
                              hipStream_t stream) {
  (void)in_sizes; (void)n_in; (void)out_size; (void)ws_size;
  const float* q  = (const float*)d_in[0];
  const float* k  = (const float*)d_in[1];
  const float* v  = (const float*)d_in[2];
  const float* wq = (const float*)d_in[3];
  const float* wk = (const float*)d_in[4];
  const float* wv = (const float*)d_in[5];
  unsigned short* ws = (unsigned short*)d_ws; // 3 * 4096 * 1024 bf16 = 24 MB
  float* out = (float*)d_out;

  dim3 pgrid(HID_ / 64, MROWS / 64, 3); // (16, 64, 3)
  proj_bf16_wmma<<<pgrid, 128, 0, stream>>>(q, k, v, wq, wk, wv, ws);

  dim3 agrid(S_ / 64, H_, B_); // (32, 16, 2)
  attn_flash_wmma<<<agrid, 128, 0, stream>>>(ws, out);
}
